// RelativeMultiHeadAttention_15685220565610
// MI455X (gfx1250) — matrix operand order repaired and checked by reading, not compiled, not run on hardware
//
#include <hip/hip_runtime.h>
#include <hip/hip_bf16.h>
#include <math.h>
#include <stdint.h>

typedef __bf16 bf16_t;
typedef __attribute__((ext_vector_type(16))) __bf16 v16bf;
typedef __attribute__((ext_vector_type(8)))  __bf16 v8bf;
typedef __attribute__((ext_vector_type(8)))  float  v8f;
typedef __attribute__((ext_vector_type(4)))  unsigned int v4u;
typedef __attribute__((ext_vector_type(8)))  int v8i;
typedef __attribute__((ext_vector_type(4)))  int v4i;

#define B_    16
#define L_    512
#define NH_   16
#define DK_   64
#define DV_   64
#define DM_   1024
#define TWO_L 1024

static __device__ __forceinline__ v8f wmma_bf16(v16bf a, v16bf b, v8f c) {
  // D = A(16x32 bf16) x B(32x16 bf16) + C(16x16 f32)
  return __builtin_amdgcn_wmma_f32_16x16x32_bf16(false, a, false, b, (short)0, c,
                                                 false, false);
}

static __device__ __forceinline__ v8f v8f_zero() {
  v8f z = {0.f, 0.f, 0.f, 0.f, 0.f, 0.f, 0.f, 0.f};
  return z;
}

// A-layout fragment (16x32 bf16): lane holds row = lane%16,
// element e -> K = e + 16*(lane>=16).
// Caller passes base already offset by (row*stride + 16*(lane>=16) [+32*kstep]);
// we load base[0..7] and base[8..15].
static __device__ __forceinline__ v16bf ld_frag_a(const bf16_t* base) {
  const v8bf* p = (const v8bf*)base;
  v8bf lo = p[0];
  v8bf hi = p[1];
  v16bf r;
#pragma unroll
  for (int i = 0; i < 8; ++i) { r[i] = lo[i]; r[i + 8] = hi[i]; }
  return r;
}

// B-layout fragment (32x16 bf16): lane holds col = lane%16,
// element e -> K = e + 16*(lane>=16). Contiguous 16 elements.
static __device__ __forceinline__ v16bf ld_frag_b(const bf16_t* base) {
  return *(const v16bf*)base;
}

// ---------------------------------------------------------------------------
// TDM: async DMA of a 64x64 bf16 tile (row pitch 64 elems) from global -> LDS.
// D# per CDNA5 ISA ch.8: group0 = {flags, lds_addr, global_addr, type=2},
// group1 = {data_size=2B, tensor_dim0=64, tensor_dim1=512, tile 64x64,
//           dim0_stride=64}; groups 2/3 zero (2-D tile). Tracked by TENSORcnt.
// This toolchain exposes the 6-arg builtin:
//   (uint32x4 g0, int32x8 g1, int32x4 g2, int32x4 g3, int32x8 g4, i32 cpol)
// ---------------------------------------------------------------------------
static __device__ __forceinline__ void tdm_load_tile_bf16_64x64(
    const bf16_t* gbase, unsigned int lds_addr) {
  uint64_t ga = (uint64_t)(uintptr_t)gbase;
  v4u g0;
  g0[0] = 1u;                                   // count=1, user descriptor
  g0[1] = lds_addr;                             // LDS byte address
  g0[2] = (unsigned int)ga;                     // global_addr[31:0]
  g0[3] = (unsigned int)(ga >> 32) | (2u << 30);  // global_addr[56:32] | type=2
  v8i g1;
  g1[0] = (int)(1u << 16);        // data_size=1 -> 2 bytes/elem
  g1[1] = (int)(64u << 16);       // tensor_dim0 = 64 (bits 79:48)
  g1[2] = (int)(512u << 16);      // tensor_dim1 = 512 (bits 111:80)
  g1[3] = (int)(64u << 16);       // tile_dim0 = 64 (bits 127:112)
  g1[4] = 64;                     // tile_dim1 = 64 (bits 143:128)
  g1[5] = 64;                     // tensor_dim0_stride = 64 (bits 207:160)
  g1[6] = 0;
  g1[7] = 0;
  v4i gz4;
  gz4[0] = 0; gz4[1] = 0; gz4[2] = 0; gz4[3] = 0;
  v8i gz8;
#pragma unroll
  for (int i = 0; i < 8; ++i) gz8[i] = 0;
  __builtin_amdgcn_tensor_load_to_lds(g0, g1, gz4, gz4, gz8, 0);
}

// ---------------------------------------------------------------------------
// Kernel 1: sinusoidal relative-position table (bf16) + r_w_bias . pos table
// ---------------------------------------------------------------------------
__global__ void pos_rw_kernel(bf16_t* __restrict__ posb, float* __restrict__ rwpos,
                              const float* __restrict__ r_w_bias) {
  int l = blockIdx.x;   // 0..1023  (relative position p = l - 512)
  int d = threadIdx.x;  // 0..63
  float p = (float)(l - L_);
  int i = d & 31;
  const float c = 9.210340372f / 31.0f;  // ln(10000)/(half-1)
  float inv = __expf(-(float)i * c);
  float ang = p * inv;
  float v = (d < 32) ? __sinf(ang) : __cosf(ang);
  posb[(size_t)l * DK_ + d] = (bf16_t)v;

  __shared__ float row[DK_];
  row[d] = v;
  __syncthreads();
  if (d < NH_) {
    float s = 0.f;
#pragma unroll
    for (int dd = 0; dd < DK_; ++dd) s += r_w_bias[d * DK_ + dd] * row[dd];
    rwpos[(size_t)d * TWO_L + l] = s;
  }
}

// ---------------------------------------------------------------------------
// Kernel 2: projection GEMM  out[b,n,l,d] = X[b,l,:] . W[n*64+d,:] + bias
// M = B*L = 8192, N = NH*DK = 1024, K = DM = 1024.  64x64 tile / 4 waves.
// ---------------------------------------------------------------------------
__global__ void __launch_bounds__(128) proj_kernel(const float* __restrict__ X,
                                                   const float* __restrict__ W,
                                                   const float* __restrict__ bias,
                                                   bf16_t* __restrict__ out) {
  int m0 = blockIdx.x * 64;
  int n0 = blockIdx.y * 64;
  int tid = threadIdx.x;
  int w = tid >> 5;
  int lane = tid & 31;
  int hv = lane >> 4;   // half-wave select
  int ln = lane & 15;

  v8f acc[4];
#pragma unroll
  for (int nb = 0; nb < 4; ++nb) acc[nb] = v8f_zero();

  int arow = m0 + w * 16 + ln;  // A row for this lane
  const float* Xr = X + (size_t)arow * DM_;
  int kbA = hv * 16;
  int kbB = hv * 16;

#pragma unroll 1
  for (int k0 = 0; k0 < DM_; k0 += 32) {
    v16bf af;
#pragma unroll
    for (int e = 0; e < 8; ++e) {
      af[e]     = (bf16_t)Xr[k0 + kbA + e];
      af[e + 8] = (bf16_t)Xr[k0 + kbA + 8 + e];
    }
#pragma unroll
    for (int nb = 0; nb < 4; ++nb) {
      int gn = n0 + nb * 16 + ln;
      const float* Wr = W + (size_t)gn * DM_ + k0 + kbB;
      v16bf bfv;
#pragma unroll
      for (int e = 0; e < 16; ++e) bfv[e] = (bf16_t)Wr[e];
      acc[nb] = wmma_bf16(af, bfv, acc[nb]);
    }
  }

#pragma unroll
  for (int nb = 0; nb < 4; ++nb) {
    int gn = n0 + nb * 16 + ln;
    int n = gn >> 6, d = gn & 63;
    float bv = bias[gn];
#pragma unroll
    for (int r = 0; r < 8; ++r) {
      int gm = m0 + w * 16 + hv * 8 + r;
      int bb = gm >> 9, ll = gm & 511;
      out[(((size_t)bb * NH_ + n) * L_ + ll) * DK_ + d] = (bf16_t)(acc[nb][r] + bv);
    }
  }
}

// ---------------------------------------------------------------------------
// Kernel 3: rrk[b,n,k] = r_r_bias[n] . K[b,n,k,:]
// ---------------------------------------------------------------------------
__global__ void rrk_kernel(const bf16_t* __restrict__ kb,
                           const float* __restrict__ r_r_bias,
                           float* __restrict__ rrk) {
  int i = blockIdx.x * blockDim.x + threadIdx.x;  // 0 .. B*NH*L-1
  int n = (i / L_) % NH_;
  float s = 0.f;
#pragma unroll
  for (int d = 0; d < DK_; ++d)
    s += (float)kb[(size_t)i * DK_ + d] * r_r_bias[n * DK_ + d];
  rrk[i] = s;
}

// ---------------------------------------------------------------------------
// Kernel 4: fused attention, flash-style streaming softmax.
//   score[q,k] = ( Q.K + rrk[k] + Qpos[q,L+k-q] + rwpos[L+k-q] + Kpos[k,L+q-k] )/8
// Block: 128 threads (4 waves). q-tile TM=64, k loop in TN=64 tiles.
// V tiles are DMA'd to LDS by the Tensor Data Mover (double-buffered).
// ---------------------------------------------------------------------------
__global__ void __launch_bounds__(128) attn_kernel(
    const bf16_t* __restrict__ qb, const bf16_t* __restrict__ kb,
    const bf16_t* __restrict__ vb, const bf16_t* __restrict__ posb,
    const float* __restrict__ rwpos, const float* __restrict__ rrk,
    const int* __restrict__ mask, float* __restrict__ out) {
  __shared__ __align__(32) bf16_t Bband_s[64][128];    // [q-local][j]  16KB
  __shared__ __align__(32) bf16_t Eband_s[64][128];    // [k-local][j]  16KB
  __shared__ __align__(32) bf16_t Ptile_s[64][64];     // probs          8KB
  __shared__ __align__(32) bf16_t Vtile_s[2][64][64];  // V dbl-buf     16KB

  int q0 = blockIdx.x * 64;
  int n  = blockIdx.y;
  int b  = blockIdx.z;
  int tid = threadIdx.x;
  int w = tid >> 5;
  int lane = tid & 31;
  int hv = lane >> 4;
  int ln = lane & 15;

  size_t bn = (size_t)b * NH_ + n;
  const bf16_t* Q = qb + bn * L_ * DK_;
  const bf16_t* K = kb + bn * L_ * DK_;
  const bf16_t* V = vb + bn * L_ * DV_;

  // persistent Q A-frags for rows q0 + w*16 + (lane%16)
  int qrow = q0 + w * 16 + ln;
  v16bf qfrag[2];
#pragma unroll
  for (int ds = 0; ds < 2; ++ds)
    qfrag[ds] = ld_frag_a(Q + (size_t)qrow * DK_ + hv * 16 + 32 * ds);

  v8f oacc[4];
  float rowmax[8], rowsum[8];
#pragma unroll
  for (int nb = 0; nb < 4; ++nb) oacc[nb] = v8f_zero();
#pragma unroll
  for (int r = 0; r < 8; ++r) { rowmax[r] = -1e30f; rowsum[r] = 0.f; }

#pragma unroll 1
  for (int k0 = 0; k0 < L_; k0 += 64) {
    int buf = (k0 >> 6) & 1;
    // ---- kick off async V-tile DMA (TDM) for this iteration
    if (w == 0) {
      tdm_load_tile_bf16_64x64(
          V + (size_t)k0 * DV_,
          (unsigned int)(uintptr_t)&Vtile_s[buf][0][0]);
    }
    // ---- prefetch next K tile (contiguous rows) into cache
    if (k0 + 64 < L_)
      __builtin_prefetch(K + (size_t)(k0 + 64) * DK_ + (size_t)tid * 64, 0, 1);

    // ---- AC = Q . K^T  (this wave: rows w*16.., all 64 cols)
    v8f acc[4];
#pragma unroll
    for (int nb = 0; nb < 4; ++nb) acc[nb] = v8f_zero();
#pragma unroll
    for (int ds = 0; ds < 2; ++ds) {
#pragma unroll
      for (int nb = 0; nb < 4; ++nb) {
        const bf16_t* kp =
            K + (size_t)(k0 + nb * 16 + ln) * DK_ + hv * 16 + 32 * ds;
        acc[nb] = wmma_bf16(qfrag[ds], ld_frag_b(kp), acc[nb]);
      }
    }

    // ---- B band: Bband[qq][j] = Q[q0+qq] . pos[lB0+j],  l = L + k - q
    int lB0 = L_ + k0 - q0 - 63;
    {
      v8f bacc[8];
#pragma unroll
      for (int jb = 0; jb < 8; ++jb) bacc[jb] = v8f_zero();
#pragma unroll
      for (int ds = 0; ds < 2; ++ds) {
#pragma unroll
        for (int jb = 0; jb < 8; ++jb) {
          int l = lB0 + jb * 16 + ln;
          l = l < 0 ? 0 : (l > TWO_L - 1 ? TWO_L - 1 : l);
          const bf16_t* pp = posb + (size_t)l * DK_ + hv * 16 + 32 * ds;
          bacc[jb] = wmma_bf16(qfrag[ds], ld_frag_b(pp), bacc[jb]);
        }
      }
#pragma unroll
      for (int jb = 0; jb < 8; ++jb)
#pragma unroll
        for (int r = 0; r < 8; ++r)
          Bband_s[w * 16 + hv * 8 + r][jb * 16 + ln] = (bf16_t)bacc[jb][r];
    }

    // ---- E band: Eband[kk][j] = K[k0+kk] . pos[lE0+j],  l = L + q - k
    int lE0 = L_ + q0 - k0 - 63;
    {
      v16bf kfrag[2];
#pragma unroll
      for (int ds = 0; ds < 2; ++ds)
        kfrag[ds] =
            ld_frag_a(K + (size_t)(k0 + w * 16 + ln) * DK_ + hv * 16 + 32 * ds);
      v8f eacc[8];
#pragma unroll
      for (int jb = 0; jb < 8; ++jb) eacc[jb] = v8f_zero();
#pragma unroll
      for (int ds = 0; ds < 2; ++ds) {
#pragma unroll
        for (int jb = 0; jb < 8; ++jb) {
          int l = lE0 + jb * 16 + ln;
          l = l < 0 ? 0 : (l > TWO_L - 1 ? TWO_L - 1 : l);
          const bf16_t* pp = posb + (size_t)l * DK_ + hv * 16 + 32 * ds;
          eacc[jb] = wmma_bf16(kfrag[ds], ld_frag_b(pp), eacc[jb]);
        }
      }
#pragma unroll
      for (int jb = 0; jb < 8; ++jb)
#pragma unroll
        for (int r = 0; r < 8; ++r)
          Eband_s[w * 16 + hv * 8 + r][jb * 16 + ln] = (bf16_t)eacc[jb][r];
    }
    __syncthreads();

    // ---- gather diagonals + pointwise terms + online softmax
    float sc[4][8];
#pragma unroll
    for (int nb = 0; nb < 4; ++nb) {
      int kk = nb * 16 + ln;
      int kg = k0 + kk;
      float rrb = rrk[bn * L_ + kg];
      int mv = mask[b * L_ + kg];
#pragma unroll
      for (int r = 0; r < 8; ++r) {
        int qq = w * 16 + hv * 8 + r;  // tile-local q row
        int qg = q0 + qq;
        float s = acc[nb][r] + rrb + rwpos[(size_t)n * TWO_L + (L_ + kg - qg)] +
                  (float)Bband_s[qq][kk - qq + 63] +
                  (float)Eband_s[kk][qq - kk + 63];
        s *= 0.125f;  // 1/sqrt(DK)
        sc[nb][r] = (mv == 0) ? -1e30f : s;
      }
    }

#pragma unroll
    for (int r = 0; r < 8; ++r) {
      float m = fmaxf(fmaxf(sc[0][r], sc[1][r]), fmaxf(sc[2][r], sc[3][r]));
#pragma unroll
      for (int off = 1; off < 16; off <<= 1)
        m = fmaxf(m, __shfl_xor(m, off, 32));
      float nm = fmaxf(rowmax[r], m);
      float scale = __expf(rowmax[r] - nm);
      rowmax[r] = nm;
      float ps = 0.f;
#pragma unroll
      for (int nb = 0; nb < 4; ++nb) {
        float p = __expf(sc[nb][r] - nm);
        sc[nb][r] = p;
        ps += p;
      }
#pragma unroll
      for (int off = 1; off < 16; off <<= 1) ps += __shfl_xor(ps, off, 32);
      rowsum[r] = rowsum[r] * scale + ps;
#pragma unroll
      for (int nb = 0; nb < 4; ++nb) oacc[nb][r] *= scale;
    }

    // probabilities -> LDS (wave-local rows) for PV A-frags
#pragma unroll
    for (int nb = 0; nb < 4; ++nb)
#pragma unroll
      for (int r = 0; r < 8; ++r)
        Ptile_s[w * 16 + hv * 8 + r][nb * 16 + ln] = (bf16_t)sc[nb][r];

    // ---- make sure the V-tile DMA has landed, then sync the workgroup
    if (w == 0) __builtin_amdgcn_s_wait_tensorcnt((short)0);
    __syncthreads();

    // ---- O += P . V   (V read from the TDM-filled LDS tile)
#pragma unroll
    for (int ks = 0; ks < 2; ++ks) {
      v16bf pa =
          ld_frag_a((const bf16_t*)&Ptile_s[w * 16 + ln][0] + hv * 16 + 32 * ks);
#pragma unroll
      for (int nb = 0; nb < 4; ++nb) {
        v16bf vf;
#pragma unroll
        for (int e = 0; e < 16; ++e)
          vf[e] = Vtile_s[buf][hv * 16 + 32 * ks + e][nb * 16 + ln];
        oacc[nb] = wmma_bf16(pa, vf, oacc[nb]);
      }
    }
  }

  // ---- epilogue: normalize and store [b, q, n*DV + dv] (f32)
#pragma unroll
  for (int nb = 0; nb < 4; ++nb) {
#pragma unroll
    for (int r = 0; r < 8; ++r) {
      int qg = q0 + w * 16 + hv * 8 + r;
      float o = oacc[nb][r] / rowsum[r];
      out[((size_t)b * L_ + qg) * (NH_ * DV_) + n * DV_ + nb * 16 + ln] = o;
    }
  }
}

// ---------------------------------------------------------------------------
extern "C" void kernel_launch(void* const* d_in, const int* in_sizes, int n_in,
                              void* d_out, int out_size, void* d_ws,
                              size_t ws_size, hipStream_t stream) {
  (void)in_sizes; (void)n_in; (void)out_size; (void)ws_size;
  const float* queries = (const float*)d_in[0];
  const float* keys    = (const float*)d_in[1];
  const float* values  = (const float*)d_in[2];
  const int*   mask    = (const int*)d_in[3];
  // d_in[4] = attn_bias (unused by reference)
  const float* Wq = (const float*)d_in[5];
  const float* bq = (const float*)d_in[6];
  const float* Wk = (const float*)d_in[7];
  const float* bk = (const float*)d_in[8];
  const float* Wv = (const float*)d_in[9];
  const float* bv = (const float*)d_in[10];
  const float* rr = (const float*)d_in[11];  // r_r_bias
  const float* rw = (const float*)d_in[12];  // r_w_bias

  char* ws = (char*)d_ws;
  size_t off = 0;
  bf16_t* qb = (bf16_t*)(ws + off); off += (size_t)B_ * NH_ * L_ * DK_ * 2;
  bf16_t* kb = (bf16_t*)(ws + off); off += (size_t)B_ * NH_ * L_ * DK_ * 2;
  bf16_t* vb = (bf16_t*)(ws + off); off += (size_t)B_ * NH_ * L_ * DV_ * 2;
  bf16_t* posb = (bf16_t*)(ws + off); off += (size_t)TWO_L * DK_ * 2;
  float* rwpos = (float*)(ws + off); off += (size_t)NH_ * TWO_L * 4;
  float* rrk   = (float*)(ws + off); off += (size_t)B_ * NH_ * L_ * 4;

  pos_rw_kernel<<<TWO_L, DK_, 0, stream>>>(posb, rwpos, rw);

  dim3 pg((B_ * L_) / 64, (NH_ * DK_) / 64);
  proj_kernel<<<pg, 128, 0, stream>>>(queries, Wq, bq, qb);
  proj_kernel<<<pg, 128, 0, stream>>>(keys,    Wk, bk, kb);
  proj_kernel<<<pg, 128, 0, stream>>>(values,  Wv, bv, vb);

  rrk_kernel<<<(B_ * NH_ * L_) / 128, 128, 0, stream>>>(kb, rr, rrk);

  attn_kernel<<<dim3(L_ / 64, NH_, B_), 128, 0, stream>>>(
      qb, kb, vb, posb, rwpos, rrk, mask, (float*)d_out);
}
